// EGT_47914655154484
// MI455X (gfx1250) — compile-verified, matrix operand order checked by reading, not run on hardware
//
#include <hip/hip_runtime.h>
#include <hip/hip_bf16.h>

// ---------------------------------------------------------------------------
// EGT (Edge-augmented Graph Transformer), 2 layers, on gfx1250 (MI455X).
// Dense matmuls use v_wmma_f32_16x16x32_bf16 (wave32 WMMA, f32 accumulate).
// Block tile 128x64, 8 waves, each wave 16x64 (4 WMMAs / K-step).
// q/k/v are stored head-major bf16 (q[z][l][d], k[z][m][d], vT[z][d][m],
// z = b*8+h) so all attention staging is contiguous b128 copies.
// ---------------------------------------------------------------------------

#define Bsz 2
#define Nn  1024
#define Ff  512
#define Ee  16
#define Hh  8
#define NVv 4
#define Ii  2048
#define Dd  64
#define Mrows (Bsz * Nn)   // 2048
#define LDK 40             // LDS row stride in ushorts (80B: 16B-aligned, conflict-free)

typedef __attribute__((ext_vector_type(16))) __bf16 v16bf;
typedef __attribute__((ext_vector_type(8)))  float  v8f;
typedef __attribute__((ext_vector_type(4)))  float  f32x4;
typedef __attribute__((ext_vector_type(4)))  unsigned int u32x4;
typedef __attribute__((ext_vector_type(2)))  unsigned int u32x2;

static __device__ __forceinline__ unsigned short f2bf(float x) {
    return __builtin_bit_cast(unsigned short, (__bf16)x);
}

static __device__ __forceinline__ v8f wmma_bf16(v16bf a, v16bf b, v8f c) {
    // (neg_a, A, neg_b, B, c_mod, C, reuse_a, reuse_b)
    return __builtin_amdgcn_wmma_f32_16x16x32_bf16(false, a, false, b, (short)0, c,
                                                   false, false);
}

// 16x32 bf16 fragment (ISA 7.12.2): lane holds row=lane%16; element e maps to
// k = (e/8)*16 + (lane/16)*8 + (e%8).  With LDK=40 both 4-dword halves are
// 16B-aligned -> two b128 LDS loads.  B uses the mirrored layout (lane%16 =
// column); B tiles are staged transposed so the same gather is reused.
static __device__ __forceinline__ v16bf load_frag(const unsigned short* base,
                                                  int row, int half) {
    const u32x4* p = (const u32x4*)(base + row * LDK + half * 8);
    union { u32x4 q[2]; v16bf v; } f;
    f.q[0] = p[0];   // dwords 0..3  -> k = half*8 + 0..7
    f.q[1] = p[2];   // dwords 8..11 -> k = 16 + half*8 + 0..7
    return f.v;
}

// ---------------------------------------------------------------------------
// LayerNorm over F=512, writes bf16. One block (256 threads) per row.
// ---------------------------------------------------------------------------
__global__ void ln_kernel(const float* __restrict__ x,
                          const float* __restrict__ g,
                          const float* __restrict__ bta,
                          __bf16* __restrict__ out) {
    const int row = blockIdx.x;
    const int tid = threadIdx.x;
    __shared__ float ssum[8], ssq[8];
    const float v0 = x[row * Ff + tid];
    const float v1 = x[row * Ff + tid + 256];
    float s = v0 + v1, sq = v0 * v0 + v1 * v1;
    for (int off = 16; off > 0; off >>= 1) {
        s  += __shfl_xor(s, off);
        sq += __shfl_xor(sq, off);
    }
    const int wid = tid >> 5, lane = tid & 31;
    if (lane == 0) { ssum[wid] = s; ssq[wid] = sq; }
    __syncthreads();
    float ts = 0.f, tq = 0.f;
    for (int w = 0; w < 8; ++w) { ts += ssum[w]; tq += ssq[w]; }
    const float mean = ts * (1.f / Ff);
    const float var  = tq * (1.f / Ff) - mean * mean;
    const float inv  = rsqrtf(var + 1e-5f);
    out[row * Ff + tid]       = (__bf16)((v0 - mean) * inv * g[tid] + bta[tid]);
    out[row * Ff + tid + 256] = (__bf16)((v1 - mean) * inv * g[tid + 256] + bta[tid + 256]);
}

// ---------------------------------------------------------------------------
// Generic tiled WMMA GEMM:  out[M,Nc] = A_bf16[M,K] * W_f32[K,Nc] + bias
//                                       (+resid) (ELU) -> f32 or bf16
// ---------------------------------------------------------------------------
template <int ACT, bool RESID, bool OUTBF>
__global__ void gemm_kernel(const __bf16* __restrict__ A,
                            const float* __restrict__ W,
                            const float* __restrict__ bias,
                            const float* __restrict__ resid,
                            float* __restrict__ outF,
                            __bf16* __restrict__ outB,
                            int Kdim, int Ncols) {
    __shared__ unsigned short As[128][LDK];
    __shared__ unsigned short Bs[64][LDK];
    const int tid  = threadIdx.x;
    const int wid  = tid >> 5, lane = tid & 31;
    const int half = lane >> 4, l16 = lane & 15;
    const int m0 = blockIdx.y * 128;
    const int n0 = blockIdx.x * 64;
    const unsigned short* Abits = (const unsigned short*)A;

    v8f acc[4] = {v8f{}, v8f{}, v8f{}, v8f{}};
    for (int k0 = 0; k0 < Kdim; k0 += 32) {
        #pragma unroll
        for (int i = 0; i < 2; ++i) {   // A tile: 128x32 bf16, b128 copies
            const int c = tid + i * 256;
            const int r = c >> 2, kk = (c & 3) * 8;
            *(u32x4*)&As[r][kk] =
                *(const u32x4*)(Abits + (size_t)(m0 + r) * Kdim + k0 + kk);
        }
        #pragma unroll
        for (int i = 0; i < 2; ++i) {   // B tile: 32x64 f32 -> bf16, transposed
            const int c = tid + i * 256;
            const int nn0 = (c & 15) * 4, kk = c >> 4;
            const f32x4 w = *(const f32x4*)(W + (size_t)(k0 + kk) * Ncols + n0 + nn0);
            Bs[nn0 + 0][kk] = f2bf(w[0]);
            Bs[nn0 + 1][kk] = f2bf(w[1]);
            Bs[nn0 + 2][kk] = f2bf(w[2]);
            Bs[nn0 + 3][kk] = f2bf(w[3]);
        }
        __syncthreads();
        const v16bf af = load_frag(&As[0][0], wid * 16 + l16, half);
        #pragma unroll
        for (int j = 0; j < 4; ++j) {
            const v16bf bfj = load_frag(&Bs[0][0], j * 16 + l16, half);
            acc[j] = wmma_bf16(af, bfj, acc[j]);
        }
        __syncthreads();
    }

    const int rb = m0 + wid * 16 + half * 8;
    #pragma unroll
    for (int j = 0; j < 4; ++j) {
        const int col = n0 + j * 16 + l16;
        const float bj = bias[col];
        #pragma unroll
        for (int r = 0; r < 8; ++r) {
            const int row = rb + r;
            float c = acc[j][r] + bj;
            if (RESID) c += resid[(size_t)row * Ncols + col];
            if (ACT == 1) c = c > 0.f ? c : expf(c) - 1.f;   // ELU
            if (OUTBF) outB[(size_t)row * Ncols + col] = (__bf16)c;
            else       outF[(size_t)row * Ncols + col] = c;
        }
    }
}

// ---------------------------------------------------------------------------
// QKV GEMM:  same 128x64 WMMA tile, but the epilogue scatters into head-major
// bf16 buffers:  q[z][l][64], k[z][m][64], vT[z][d][1024]  (z = b*8+h).
// Column c of the 1536-wide output maps to (dd = c/8, h = c%8).
// ---------------------------------------------------------------------------
__global__ void gemm_qkv_kernel(const __bf16* __restrict__ A,
                                const float* __restrict__ W,
                                const float* __restrict__ bias,
                                __bf16* __restrict__ qbuf,
                                __bf16* __restrict__ kbuf,
                                __bf16* __restrict__ vtbuf) {
    const int Kdim = Ff, Ncols = 3 * Ff;
    __shared__ unsigned short As[128][LDK];
    __shared__ unsigned short Bs[64][LDK];
    const int tid  = threadIdx.x;
    const int wid  = tid >> 5, lane = tid & 31;
    const int half = lane >> 4, l16 = lane & 15;
    const int m0 = blockIdx.y * 128;
    const int n0 = blockIdx.x * 64;
    const unsigned short* Abits = (const unsigned short*)A;

    v8f acc[4] = {v8f{}, v8f{}, v8f{}, v8f{}};
    for (int k0 = 0; k0 < Kdim; k0 += 32) {
        #pragma unroll
        for (int i = 0; i < 2; ++i) {
            const int c = tid + i * 256;
            const int r = c >> 2, kk = (c & 3) * 8;
            *(u32x4*)&As[r][kk] =
                *(const u32x4*)(Abits + (size_t)(m0 + r) * Kdim + k0 + kk);
        }
        #pragma unroll
        for (int i = 0; i < 2; ++i) {
            const int c = tid + i * 256;
            const int nn0 = (c & 15) * 4, kk = c >> 4;
            const f32x4 w = *(const f32x4*)(W + (size_t)(k0 + kk) * Ncols + n0 + nn0);
            Bs[nn0 + 0][kk] = f2bf(w[0]);
            Bs[nn0 + 1][kk] = f2bf(w[1]);
            Bs[nn0 + 2][kk] = f2bf(w[2]);
            Bs[nn0 + 3][kk] = f2bf(w[3]);
        }
        __syncthreads();
        const v16bf af = load_frag(&As[0][0], wid * 16 + l16, half);
        #pragma unroll
        for (int j = 0; j < 4; ++j) {
            const v16bf bfj = load_frag(&Bs[0][0], j * 16 + l16, half);
            acc[j] = wmma_bf16(af, bfj, acc[j]);
        }
        __syncthreads();
    }

    const int rb = m0 + wid * 16 + half * 8;
    #pragma unroll
    for (int j = 0; j < 4; ++j) {
        const int col = n0 + j * 16 + l16;
        const int h  = col & 7;
        const int dd = col >> 3;            // 0..191
        const float bj = bias[col];
        #pragma unroll
        for (int r = 0; r < 8; ++r) {
            const int row = rb + r;         // b*1024 + l
            const int b = row >> 10, l = row & 1023;
            const int z = (b << 3) | h;
            const __bf16 v = (__bf16)(acc[j][r] + bj);
            if (dd < 64)
                qbuf[((size_t)z * Nn + l) * Dd + dd] = v;
            else if (dd < 128)
                kbuf[((size_t)z * Nn + l) * Dd + (dd - 64)] = v;
            else
                vtbuf[((size_t)z * Dd + (dd - 128)) * Nn + l] = v;   // l-contiguous
        }
    }
}

// ---------------------------------------------------------------------------
// Attention scores per z=(b,h):  S[l,m] = sum_d q[z][l][d]*k[z][m][d]
// All staging is contiguous b128 bf16 copies.
// ---------------------------------------------------------------------------
__global__ void scores_kernel(const __bf16* __restrict__ qbuf,
                              const __bf16* __restrict__ kbuf,
                              float* __restrict__ T) {
    __shared__ unsigned short As[128][LDK];
    __shared__ unsigned short Bs[64][LDK];
    const int z = blockIdx.z;            // b*8 + h
    const int l0 = blockIdx.y * 128;
    const int m0 = blockIdx.x * 64;
    const int tid  = threadIdx.x;
    const int wid  = tid >> 5, lane = tid & 31;
    const int half = lane >> 4, l16 = lane & 15;
    const unsigned short* qb = (const unsigned short*)qbuf + (size_t)z * Nn * Dd;
    const unsigned short* kb = (const unsigned short*)kbuf + (size_t)z * Nn * Dd;

    v8f acc[4] = {v8f{}, v8f{}, v8f{}, v8f{}};
    for (int k0 = 0; k0 < Dd; k0 += 32) {
        #pragma unroll
        for (int i = 0; i < 2; ++i) {   // A: 128x32 bf16, b128 copies
            const int c = tid + i * 256;
            const int r = c >> 2, kk = (c & 3) * 8;
            *(u32x4*)&As[r][kk] = *(const u32x4*)(qb + (size_t)(l0 + r) * Dd + k0 + kk);
        }
        {                               // B: 64x32 bf16, b128 copies
            const int nn = tid >> 2, kk = (tid & 3) * 8;
            *(u32x4*)&Bs[nn][kk] = *(const u32x4*)(kb + (size_t)(m0 + nn) * Dd + k0 + kk);
        }
        __syncthreads();
        const v16bf af = load_frag(&As[0][0], wid * 16 + l16, half);
        #pragma unroll
        for (int j = 0; j < 4; ++j) {
            const v16bf bfj = load_frag(&Bs[0][0], j * 16 + l16, half);
            acc[j] = wmma_bf16(af, bfj, acc[j]);
        }
        __syncthreads();
    }
    const int rb = l0 + wid * 16 + half * 8;
    #pragma unroll
    for (int j = 0; j < 4; ++j) {
        const int col = m0 + j * 16 + l16;
        #pragma unroll
        for (int r = 0; r < 8; ++r)
            T[((size_t)z * Nn + (rb + r)) * Nn + col] = acc[j][r];
    }
}

// ---------------------------------------------------------------------------
// Fused e_bias/gates/clip/softmax/scalers. One block (256 thr) per (b,l).
// ---------------------------------------------------------------------------
__global__ void attn_softmax_kernel(const float* __restrict__ efeat,
                                    float* __restrict__ T,
                                    float* __restrict__ Ahat,
                                    const float* __restrict__ We,
                                    const float* __restrict__ be,
                                    const float* __restrict__ Wg,
                                    const float* __restrict__ bg) {
    const int b = blockIdx.x >> 10;
    const int l = blockIdx.x & 1023;
    const int tid = threadIdx.x;
    const int wid = tid >> 5, lane = tid & 31;
    __shared__ float sWe[16][8], sWg[16][8], sbe[8], sbg[8];
    __shared__ float red[8][8], redg[8][8];
    if (tid < 128) { sWe[tid >> 3][tid & 7] = We[tid]; sWg[tid >> 3][tid & 7] = Wg[tid]; }
    if (tid < 8)   { sbe[tid] = be[tid]; sbg[tid] = bg[tid]; }
    __syncthreads();

    float t_loc[4][8], g_loc[4][8];
    float mx[8], gs[8];
    #pragma unroll
    for (int h = 0; h < 8; ++h) { mx[h] = -1e30f; gs[h] = 0.f; }
    const float* erow = efeat + ((size_t)(b * Nn + l)) * Nn * Ee;

    #pragma unroll
    for (int i = 0; i < 4; ++i) {
        const int m = tid + i * 256;
        const f32x4* e4 = (const f32x4*)(erow + (size_t)m * Ee);
        f32x4 q[4];
        q[0] = e4[0]; q[1] = e4[1]; q[2] = e4[2]; q[3] = e4[3];
        float ef[16];
        #pragma unroll
        for (int e = 0; e < 16; ++e) ef[e] = q[e >> 2][e & 3];
        #pragma unroll
        for (int h = 0; h < 8; ++h) {
            float bias = sbe[h], gg = sbg[h];
            #pragma unroll
            for (int e = 0; e < 16; ++e) {
                bias += ef[e] * sWe[e][h];
                gg   += ef[e] * sWg[e][h];
            }
            const float gate = 1.f / (1.f + expf(-gg));
            const float s = T[(((size_t)(b * 8 + h) * Nn) + l) * Nn + m];
            const float t = fminf(fmaxf(s, -5.f), 5.f) + bias;
            t_loc[i][h] = t; g_loc[i][h] = gate;
            mx[h] = fmaxf(mx[h], t);
            gs[h] += gate;
        }
    }
    #pragma unroll
    for (int h = 0; h < 8; ++h)
        for (int off = 16; off > 0; off >>= 1) {
            mx[h] = fmaxf(mx[h], __shfl_xor(mx[h], off));
            gs[h] += __shfl_xor(gs[h], off);
        }
    if (lane == 0) {
        #pragma unroll
        for (int h = 0; h < 8; ++h) { red[wid][h] = mx[h]; redg[wid][h] = gs[h]; }
    }
    __syncthreads();
    #pragma unroll
    for (int h = 0; h < 8; ++h) {
        float M = -1e30f, G = 0.f;
        for (int w = 0; w < 8; ++w) { M = fmaxf(M, red[w][h]); G += redg[w][h]; }
        mx[h] = M; gs[h] = G;
    }
    __syncthreads();

    float es[8];
    #pragma unroll
    for (int h = 0; h < 8; ++h) es[h] = 0.f;
    #pragma unroll
    for (int i = 0; i < 4; ++i) {
        #pragma unroll
        for (int h = 0; h < 8; ++h) es[h] += expf(t_loc[i][h] - mx[h]);
    }
    #pragma unroll
    for (int h = 0; h < 8; ++h)
        for (int off = 16; off > 0; off >>= 1) es[h] += __shfl_xor(es[h], off);
    if (lane == 0) {
        #pragma unroll
        for (int h = 0; h < 8; ++h) red[wid][h] = es[h];
    }
    __syncthreads();
    #pragma unroll
    for (int h = 0; h < 8; ++h) {
        float S = 0.f;
        for (int w = 0; w < 8; ++w) S += red[w][h];
        es[h] = 1.f / S;
    }
    float scal[8];
    #pragma unroll
    for (int h = 0; h < 8; ++h) scal[h] = (l < NVv) ? 1.f : log1pf(gs[h]);

    #pragma unroll
    for (int i = 0; i < 4; ++i) {
        const int m = tid + i * 256;
        f32x4 a0, a1;
        #pragma unroll
        for (int h = 0; h < 8; ++h) {
            const float ah = expf(t_loc[i][h] - mx[h]) * es[h];
            if (h < 4) a0[h] = ah; else a1[h - 4] = ah;
            T[(((size_t)(b * 8 + h) * Nn) + l) * Nn + m] = ah * g_loc[i][h] * scal[h];
        }
        f32x4* o = (f32x4*)(Ahat + (((size_t)(b * Nn + l)) * Nn + m) * 8);
        o[0] = a0; o[1] = a1;
    }
}

// ---------------------------------------------------------------------------
// AV per z=(b,h):  vattn[b,l, d*8+h] = sum_m T[z,l,m] * vT[z][d][m]
// ---------------------------------------------------------------------------
__global__ void av_kernel(const float* __restrict__ T,
                          const __bf16* __restrict__ vtbuf,
                          __bf16* __restrict__ vattn) {
    __shared__ unsigned short As[128][LDK];
    __shared__ unsigned short Bs[64][LDK];
    const int z = blockIdx.z;
    const int b = z >> 3, h = z & 7;
    const int l0 = blockIdx.y * 128;
    const int tid  = threadIdx.x;
    const int wid  = tid >> 5, lane = tid & 31;
    const int half = lane >> 4, l16 = lane & 15;
    const unsigned short* vt = (const unsigned short*)vtbuf + (size_t)z * Dd * Nn;

    v8f acc[4] = {v8f{}, v8f{}, v8f{}, v8f{}};
    for (int k0 = 0; k0 < Nn; k0 += 32) {
        #pragma unroll
        for (int i = 0; i < 4; ++i) {   // A: 128x32 from contiguous T rows, f32x4
            const int c = tid + i * 256;
            const int r = c >> 3, kk0 = (c & 7) * 4;
            const f32x4 w =
                *(const f32x4*)(T + ((size_t)z * Nn + l0 + r) * Nn + k0 + kk0);
            const unsigned int p0 =
                (unsigned)f2bf(w[0]) | ((unsigned)f2bf(w[1]) << 16);
            const unsigned int p1 =
                (unsigned)f2bf(w[2]) | ((unsigned)f2bf(w[3]) << 16);
            u32x2 pk; pk[0] = p0; pk[1] = p1;
            *(u32x2*)&As[r][kk0] = pk;
        }
        {                               // B: 64x32 bf16, b128 copies (vT rows)
            const int nn = tid >> 2, kk = (tid & 3) * 8;   // nn = d
            *(u32x4*)&Bs[nn][kk] = *(const u32x4*)(vt + (size_t)nn * Nn + k0 + kk);
        }
        __syncthreads();
        const v16bf af = load_frag(&As[0][0], wid * 16 + l16, half);
        #pragma unroll
        for (int j = 0; j < 4; ++j) {
            const v16bf bfj = load_frag(&Bs[0][0], j * 16 + l16, half);
            acc[j] = wmma_bf16(af, bfj, acc[j]);
        }
        __syncthreads();
    }
    const int rb = l0 + wid * 16 + half * 8;
    #pragma unroll
    for (int j = 0; j < 4; ++j) {
        const int d = j * 16 + l16;
        #pragma unroll
        for (int r = 0; r < 8; ++r) {
            const size_t row = (size_t)b * Nn + rb + r;
            vattn[row * Ff + d * 8 + h] = (__bf16)acc[j][r];
        }
    }
}

// ---------------------------------------------------------------------------
// Fused edge update: one thread per edge.
// ---------------------------------------------------------------------------
__global__ void edge_kernel(const float* __restrict__ Ahat,
                            float* __restrict__ efeat,
                            const float* __restrict__ Weo,
                            const float* __restrict__ beo,
                            const float* __restrict__ g,
                            const float* __restrict__ bta,
                            const float* __restrict__ We1,
                            const float* __restrict__ be1,
                            const float* __restrict__ We2,
                            const float* __restrict__ be2) {
    __shared__ float sWeo[8][16], sWe1[16][16], sWe2[16][16];
    __shared__ float sbeo[16], sg[16], sb[16], sb1[16], sb2[16];
    const int tid = threadIdx.x;
    if (tid < 128) sWeo[tid >> 4][tid & 15] = Weo[tid];
    sWe1[tid >> 4][tid & 15] = We1[tid];
    sWe2[tid >> 4][tid & 15] = We2[tid];
    if (tid < 16) {
        sbeo[tid] = beo[tid]; sg[tid] = g[tid]; sb[tid] = bta[tid];
        sb1[tid] = be1[tid];  sb2[tid] = be2[tid];
    }
    __syncthreads();

    const size_t eid = (size_t)blockIdx.x * 256 + tid;
    const f32x4* a4 = (const f32x4*)(Ahat + eid * 8);
    const f32x4 av0 = a4[0], av1 = a4[1];
    float a[8];
    #pragma unroll
    for (int h = 0; h < 8; ++h) a[h] = (h < 4) ? av0[h] : av1[h - 4];

    f32x4* erow = (f32x4*)(efeat + eid * 16);
    f32x4 eo[4];
    eo[0] = erow[0]; eo[1] = erow[1]; eo[2] = erow[2]; eo[3] = erow[3];

    float t[16];
    #pragma unroll
    for (int e = 0; e < 16; ++e) {
        float te = sbeo[e] + eo[e >> 2][e & 3];
        #pragma unroll
        for (int h = 0; h < 8; ++h) te += a[h] * sWeo[h][e];
        t[e] = te;
    }
    float mean = 0.f;
    #pragma unroll
    for (int e = 0; e < 16; ++e) mean += t[e];
    mean *= (1.f / 16.f);
    float var = 0.f;
    #pragma unroll
    for (int e = 0; e < 16; ++e) { const float d = t[e] - mean; var += d * d; }
    var *= (1.f / 16.f);
    const float inv = rsqrtf(var + 1e-5f);
    float u[16];
    #pragma unroll
    for (int e = 0; e < 16; ++e) u[e] = (t[e] - mean) * inv * sg[e] + sb[e];
    float h1[16];
    #pragma unroll
    for (int j = 0; j < 16; ++j) {
        float s = sb1[j];
        #pragma unroll
        for (int e = 0; e < 16; ++e) s += u[e] * sWe1[e][j];
        h1[j] = s > 0.f ? s : expf(s) - 1.f;
    }
    f32x4 o[4];
    #pragma unroll
    for (int e2 = 0; e2 < 16; ++e2) {
        float s = sb2[e2];
        #pragma unroll
        for (int j = 0; j < 16; ++j) s += h1[j] * sWe2[j][e2];
        o[e2 >> 2][e2 & 3] = s + t[e2];
    }
    erow[0] = o[0]; erow[1] = o[1]; erow[2] = o[2]; erow[3] = o[3];
}

// ---------------------------------------------------------------------------
// Host side
// ---------------------------------------------------------------------------
extern "C" void kernel_launch(void* const* d_in, const int* in_sizes, int n_in,
                              void* d_out, int out_size, void* d_ws, size_t ws_size,
                              hipStream_t stream) {
    (void)in_sizes; (void)n_in; (void)out_size; (void)ws_size;

    const float* in_nfeat = (const float*)d_in[0];
    const float* in_efeat = (const float*)d_in[1];
    // params in setup_inputs() dict insertion order
    const float* p_ln_h_g = (const float*)d_in[2];
    const float* p_ln_h_b = (const float*)d_in[3];
    const float* p_We     = (const float*)d_in[4];
    const float* p_be     = (const float*)d_in[5];
    const float* p_Wg     = (const float*)d_in[6];
    const float* p_bg     = (const float*)d_in[7];
    const float* p_Wqkv   = (const float*)d_in[8];
    const float* p_bqkv   = (const float*)d_in[9];
    const float* p_Wo     = (const float*)d_in[10];
    const float* p_bo     = (const float*)d_in[11];
    const float* p_ffng   = (const float*)d_in[12];
    const float* p_ffnb   = (const float*)d_in[13];
    const float* p_W1     = (const float*)d_in[14];
    const float* p_b1     = (const float*)d_in[15];
    const float* p_W2     = (const float*)d_in[16];
    const float* p_b2     = (const float*)d_in[17];
    const float* p_Weo    = (const float*)d_in[18];
    const float* p_beo    = (const float*)d_in[19];
    const float* p_eln_g  = (const float*)d_in[20];
    const float* p_eln_b  = (const float*)d_in[21];
    const float* p_We1    = (const float*)d_in[22];
    const float* p_be1    = (const float*)d_in[23];
    const float* p_We2    = (const float*)d_in[24];
    const float* p_be2    = (const float*)d_in[25];

    float* nf = (float*)d_out;                 // (B,N,F)
    float* ef = nf + (size_t)Bsz * Nn * Ff;    // (B,N,N,E)
    const size_t nfeat_bytes = (size_t)Bsz * Nn * Ff * 4;
    const size_t efeat_bytes = (size_t)Bsz * Nn * Nn * Ee * 4;

    char* ws = (char*)d_ws;
    float*  T      = (float*)(ws);              // 67.1 MB  scores / attn_tild
    float*  Ahat   = (float*)(ws + 67108864);   // 67.1 MB  attn_hat (b,l,m,h)
    __bf16* qbuf   = (__bf16*)(ws + 134217728); //  2.1 MB  q[z][l][64]
    __bf16* kbuf   = (__bf16*)(ws + 136314880); //  2.1 MB  k[z][m][64]
    __bf16* vtbuf  = (__bf16*)(ws + 138412032); //  2.1 MB  vT[z][d][1024]
    __bf16* hln    = (__bf16*)(ws + 140509184); //  2.1 MB
    __bf16* vattn  = (__bf16*)(ws + 142606336); //  2.1 MB
    __bf16* hidden = (__bf16*)(ws + 144703488); //  8.4 MB

    hipMemcpyAsync(nf, in_nfeat, nfeat_bytes, hipMemcpyDeviceToDevice, stream);
    hipMemcpyAsync(ef, in_efeat, efeat_bytes, hipMemcpyDeviceToDevice, stream);

    const dim3 blk(256);
    for (int l = 0; l < 2; ++l) {
        const float* We   = p_We   + (size_t)l * Ee * Hh;
        const float* be   = p_be   + (size_t)l * Hh;
        const float* Wg   = p_Wg   + (size_t)l * Ee * Hh;
        const float* bg   = p_bg   + (size_t)l * Hh;
        const float* Wqkv = p_Wqkv + (size_t)l * Ff * 3 * Ff;
        const float* bqkv = p_bqkv + (size_t)l * 3 * Ff;
        const float* Wo   = p_Wo   + (size_t)l * Ff * Ff;
        const float* bo   = p_bo   + (size_t)l * Ff;
        const float* W1   = p_W1   + (size_t)l * Ff * Ii;
        const float* b1   = p_b1   + (size_t)l * Ii;
        const float* W2   = p_W2   + (size_t)l * Ii * Ff;
        const float* b2   = p_b2   + (size_t)l * Ff;
        const float* Weo  = p_Weo  + (size_t)l * Hh * Ee;
        const float* beo  = p_beo  + (size_t)l * Ee;
        const float* elng = p_eln_g + (size_t)l * Ee;
        const float* elnb = p_eln_b + (size_t)l * Ee;
        const float* We1  = p_We1  + (size_t)l * Ee * Ee;
        const float* be1  = p_be1  + (size_t)l * Ee;
        const float* We2  = p_We2  + (size_t)l * Ee * Ee;
        const float* be2  = p_be2  + (size_t)l * Ee;

        ln_kernel<<<Mrows, blk, 0, stream>>>(nf, p_ln_h_g + (size_t)l * Ff,
                                             p_ln_h_b + (size_t)l * Ff, hln);
        gemm_qkv_kernel<<<dim3(1536 / 64, Mrows / 128), blk, 0, stream>>>(
            hln, Wqkv, bqkv, qbuf, kbuf, vtbuf);
        scores_kernel<<<dim3(Nn / 64, Nn / 128, Bsz * Hh), blk, 0, stream>>>(
            qbuf, kbuf, T);
        attn_softmax_kernel<<<Bsz * Nn, blk, 0, stream>>>(ef, T, Ahat, We, be, Wg, bg);
        av_kernel<<<dim3(1, Nn / 128, Bsz * Hh), blk, 0, stream>>>(T, vtbuf, vattn);
        gemm_kernel<0, true, false><<<dim3(Ff / 64, Mrows / 128), blk, 0, stream>>>(
            vattn, Wo, bo, nf, nf, nullptr, Ff, Ff);
        ln_kernel<<<Mrows, blk, 0, stream>>>(nf, p_ffng + (size_t)l * Ff,
                                             p_ffnb + (size_t)l * Ff, hln);
        gemm_kernel<1, false, true><<<dim3(Ii / 64, Mrows / 128), blk, 0, stream>>>(
            hln, W1, b1, nullptr, nullptr, hidden, Ff, Ii);
        gemm_kernel<0, true, false><<<dim3(Ff / 64, Mrows / 128), blk, 0, stream>>>(
            hidden, W2, b2, nf, nf, nullptr, Ii, Ff);
        edge_kernel<<<(Bsz * Nn * Nn) / 256, blk, 0, stream>>>(
            Ahat, ef, Weo, beo, elng, elnb, We1, be1, We2, be2);
    }
}
